// ParallelSelfAttention_84585085927628
// MI455X (gfx1250) — compile-verified, matrix operand order checked by reading, not run on hardware
//
#include <hip/hip_runtime.h>

// ---------------------------------------------------------------------------
// Causal multi-head self-attention for MI455X (gfx1250, wave32, WMMA).
// B=4, S=2048, HID=1024, NH=16, HD=64.
// Mixed precision: bf16 operands -> v_wmma_f32_16x16x32_bf16, f32 accumulate,
// f32 softmax. Flash-attention online softmax, 32 query rows per wave.
// GEMMs use 1x4 register blocking; all fragment loads are hoisted ahead of
// the WMMA group so loads issue as one clause and waits can be staggered.
// ---------------------------------------------------------------------------

typedef __attribute__((ext_vector_type(16))) __bf16 v16bf;
typedef __attribute__((ext_vector_type(8)))  __bf16 v8bf;
typedef __attribute__((ext_vector_type(8)))  float  v8f;

#define B_    4
#define S_    2048
#define HID_  1024
#define NH_   16
#define HD_   64
#define TOK_  (B_ * S_)        // 8192 token rows
#define NQKV_ (3 * HID_)       // 3072
#define SCALE_ 0.125f          // 1/sqrt(64)

// ---- helpers --------------------------------------------------------------

__device__ __forceinline__ unsigned short f32_to_bf16(float f) {
  unsigned int u = __builtin_bit_cast(unsigned int, f);
  u += 0x7FFFu + ((u >> 16) & 1u);   // round-to-nearest-even
  return (unsigned short)(u >> 16);
}

// Build a 16-element bf16 fragment from two contiguous 8-element (16B) chunks.
__device__ __forceinline__ v16bf load_frag(const unsigned short* p0,
                                           const unsigned short* p1) {
  v8bf lo = *(const v8bf*)p0;
  v8bf hi = *(const v8bf*)p1;
  return __builtin_shufflevector(lo, hi, 0, 1, 2, 3, 4, 5, 6, 7,
                                         8, 9, 10, 11, 12, 13, 14, 15);
}

__device__ __forceinline__ v8f wmma_bf16(v16bf a, v16bf b, v8f c) {
  return __builtin_amdgcn_wmma_f32_16x16x32_bf16(
      /*neg_a=*/false, a, /*neg_b=*/false, b,
      /*c_mod=*/(short)0, c, /*reuse_a=*/false, /*reuse_b=*/false);
}

__device__ __forceinline__ float red16_max(float v) {
  // reduce across the 16-lane group (xor masks < 16 keep halves separate)
  v = fmaxf(v, __shfl_xor(v, 1, 32));
  v = fmaxf(v, __shfl_xor(v, 2, 32));
  v = fmaxf(v, __shfl_xor(v, 4, 32));
  v = fmaxf(v, __shfl_xor(v, 8, 32));
  return v;
}

__device__ __forceinline__ float red16_sum(float v) {
  v += __shfl_xor(v, 1, 32);
  v += __shfl_xor(v, 2, 32);
  v += __shfl_xor(v, 4, 32);
  v += __shfl_xor(v, 8, 32);
  return v;
}

// ---- kernel 1: f32 -> bf16 convert ----------------------------------------

__global__ __launch_bounds__(256) void k_convert_bf16(const float* __restrict__ src,
                                                      unsigned short* __restrict__ dst,
                                                      int n) {
  int i = blockIdx.x * 256 + threadIdx.x;
  if (i < n) dst[i] = f32_to_bf16(src[i]);
}

// ---- kernel 2: pack weight [K,N] f32 -> bf16 blocked [(K/32), N, 32] -------

__global__ __launch_bounds__(256) void k_pack_weight(const float* __restrict__ w,
                                                     unsigned short* __restrict__ p,
                                                     int K, int N) {
  int i = blockIdx.x * 256 + threadIdx.x;
  if (i >= K * N) return;
  int k = i / N, n = i - k * N;
  p[(((size_t)(k >> 5) * N + n) << 5) + (k & 31)] = f32_to_bf16(w[i]);
}

// ---- kernel 3: QKV GEMM (16x64 tile per wave), scatter Q,K [BH,S,64],
// ----           V^T [BH,64,S] -----------------------------------------------

__global__ __launch_bounds__(256) void k_gemm_qkv(const unsigned short* __restrict__ X,
                                                  const unsigned short* __restrict__ Wp,
                                                  const float* __restrict__ bias,
                                                  unsigned short* __restrict__ Q,
                                                  unsigned short* __restrict__ Kmat,
                                                  unsigned short* __restrict__ Vt) {
  const int lane = threadIdx.x & 31;
  const int wave = threadIdx.x >> 5;
  const int tid  = blockIdx.x * 8 + wave;      // one 16x64 tile per wave
  const int mt   = tid / (NQKV_ / 64);         // 0..511
  const int ntg  = tid % (NQKV_ / 64);         // 0..47
  const int hv   = lane >> 4;                  // half-wave
  const int ln   = lane & 15;

  const unsigned short* arow = X + (size_t)(mt * 16 + ln) * HID_;

  v8f acc0 = {}, acc1 = {}, acc2 = {}, acc3 = {};
#pragma unroll 2
  for (int kb = 0; kb < HID_ / 32; ++kb) {
    const int k0 = kb * 32 + hv * 8;
    const unsigned short* bb =
        Wp + (((size_t)kb * NQKV_ + ntg * 64 + ln) << 5) + hv * 8;
    // hoist all loads into distinct registers so the 10 global_load_b128 can
    // issue as one clause and the compiler staggers s_wait_loadcnt
    v16bf a  = load_frag(arow + k0,      arow + k0 + 16);
    v16bf b0 = load_frag(bb,             bb + 16);
    v16bf b1 = load_frag(bb + 16 * 32,   bb + 16 * 32 + 16);
    v16bf b2 = load_frag(bb + 32 * 32,   bb + 32 * 32 + 16);
    v16bf b3 = load_frag(bb + 48 * 32,   bb + 48 * 32 + 16);
    if (kb + 1 < HID_ / 32) {
      __builtin_prefetch(arow + k0 + 32, 0, 0);
      __builtin_prefetch(bb + (NQKV_ << 5), 0, 0);
    }
    acc0 = wmma_bf16(a, b0, acc0);
    acc1 = wmma_bf16(a, b1, acc1);
    acc2 = wmma_bf16(a, b2, acc2);
    acc3 = wmma_bf16(a, b3, acc3);
  }

  v8f* accs[4] = {&acc0, &acc1, &acc2, &acc3};
#pragma unroll
  for (int t = 0; t < 4; ++t) {
    const int n  = ntg * 64 + t * 16 + ln;     // output column
    const float bv = bias[n];
    const int h  = n / (3 * HD_);
    const int rr = n - h * (3 * HD_);
#pragma unroll
    for (int r = 0; r < 8; ++r) {
      const int m = r + hv * 8;
      const int tr = mt * 16 + m;              // token row
      const int b  = tr >> 11;                 // /2048
      const int s  = tr & (S_ - 1);
      const int bh = b * NH_ + h;
      const unsigned short val = f32_to_bf16((*accs[t])[r] + bv);
      if (rr < HD_) {
        Q[((size_t)bh * S_ + s) * HD_ + rr] = val;
      } else if (rr < 2 * HD_) {
        Kmat[((size_t)bh * S_ + s) * HD_ + (rr - HD_)] = val;
      } else {
        Vt[((size_t)bh * HD_ + (rr - 2 * HD_)) * S_ + s] = val;
      }
    }
  }
}

// ---- kernel 4: flash attention, one wave per (b,h,32-query-row block) ------

__global__ __launch_bounds__(256) void k_attn(const unsigned short* __restrict__ Q,
                                              const unsigned short* __restrict__ Kmat,
                                              const unsigned short* __restrict__ Vt,
                                              unsigned short* __restrict__ Ctx) {
  __shared__ float sP[8][16][32];              // per-wave probs tile (16KB)

  const int lane = threadIdx.x & 31;
  const int wave = threadIdx.x >> 5;
  const int tid  = blockIdx.x * 8 + wave;
  const int bh   = tid >> 6;                   // 0..63
  const int qb   = tid & 63;                   // 32-row query block
  const int qbase = qb * 32;
  const int hv = lane >> 4, ln = lane & 15;
  const int b = bh >> 4, h = bh & 15;

  // Q fragments: 2 q-tiles x 2 K=32 fragments
  v16bf qf[2][2];
#pragma unroll
  for (int qt = 0; qt < 2; ++qt) {
    const unsigned short* qp = Q + ((size_t)bh * S_ + qbase + qt * 16 + ln) * HD_;
    qf[qt][0] = load_frag(qp + hv * 8,      qp + 16 + hv * 8);
    qf[qt][1] = load_frag(qp + 32 + hv * 8, qp + 48 + hv * 8);
  }

  float mi[2][8], li[2][8];
#pragma unroll
  for (int qt = 0; qt < 2; ++qt)
#pragma unroll
    for (int r = 0; r < 8; ++r) { mi[qt][r] = -1e30f; li[qt][r] = 0.0f; }
  v8f acc[2][4] = {};

  for (int kvb = 0; kvb < qbase + 32; kvb += 32) {
    // ---- shared K fragments for kv [kvb, kvb+32): 2 kv-tiles x 2 d-halves --
    const unsigned short* kp0 = Kmat + ((size_t)bh * S_ + kvb + ln) * HD_;
    const unsigned short* kp1 = Kmat + ((size_t)bh * S_ + kvb + 16 + ln) * HD_;
    v16bf kf0a = load_frag(kp0 + hv * 8,      kp0 + 16 + hv * 8);
    v16bf kf0b = load_frag(kp0 + 32 + hv * 8, kp0 + 48 + hv * 8);
    v16bf kf1a = load_frag(kp1 + hv * 8,      kp1 + 16 + hv * 8);
    v16bf kf1b = load_frag(kp1 + 32 + hv * 8, kp1 + 48 + hv * 8);

    // ---- shared V^T fragments: 4 d-tiles of 16 (contiguous kv loads) -------
    v16bf vf[4];
#pragma unroll
    for (int d = 0; d < 4; ++d) {
      const unsigned short* vp =
          Vt + ((size_t)bh * HD_ + d * 16 + ln) * S_ + kvb;
      vf[d] = load_frag(vp + hv * 8, vp + 16 + hv * 8);
    }

#pragma unroll
    for (int qt = 0; qt < 2; ++qt) {
      // skip q-tiles entirely above the causal diagonal (wave-uniform branch)
      if (kvb >= qbase + qt * 16 + 16) continue;

      // scores: two 16x16 tiles
      v8f s0 = {}, s1 = {};
      s0 = wmma_bf16(qf[qt][0], kf0a, s0);
      s0 = wmma_bf16(qf[qt][1], kf0b, s0);
      s1 = wmma_bf16(qf[qt][0], kf1a, s1);
      s1 = wmma_bf16(qf[qt][1], kf1b, s1);

      // online softmax update (C layout: N=ln, M=r+8*hv)
#pragma unroll
      for (int r = 0; r < 8; ++r) {
        const int qg  = qbase + qt * 16 + r + hv * 8;
        const int kg0 = kvb + ln;
        const int kg1 = kvb + 16 + ln;
        const float v0 = (kg0 <= qg) ? s0[r] * SCALE_ : -1e30f;
        const float v1 = (kg1 <= qg) ? s1[r] * SCALE_ : -1e30f;
        float tmax = red16_max(fmaxf(v0, v1));
        const float newm = fmaxf(mi[qt][r], tmax);
        const float p0 = __expf(v0 - newm);
        const float p1 = __expf(v1 - newm);
        sP[wave][r + hv * 8][ln]      = p0;
        sP[wave][r + hv * 8][16 + ln] = p1;
        const float rsum  = red16_sum(p0 + p1);
        const float alpha = __expf(mi[qt][r] - newm);
        li[qt][r] = li[qt][r] * alpha + rsum;
        mi[qt][r] = newm;
#pragma unroll
        for (int d = 0; d < 4; ++d) acc[qt][d][r] *= alpha;
      }

      // re-layout probs C->A fragment via wave-private LDS (in-order DS)
      union { unsigned short u[16]; v16bf v; } pa;
#pragma unroll
      for (int j = 0; j < 16; ++j) {
        const int kv = j + ((j >= 8) ? 8 : 0) + hv * 8;
        pa.u[j] = f32_to_bf16(sP[wave][ln][kv]);
      }

      // context accumulation: 4 d-tiles of 16
#pragma unroll
      for (int d = 0; d < 4; ++d) acc[qt][d] = wmma_bf16(pa.v, vf[d], acc[qt][d]);
    }
  }

  // ---- normalize and write context in [B,S,H*D] row-major (bf16) ----------
#pragma unroll
  for (int qt = 0; qt < 2; ++qt) {
#pragma unroll
    for (int r = 0; r < 8; ++r) {
      const float inv = 1.0f / li[qt][r];
      const size_t row =
          ((size_t)(b * S_ + qbase + qt * 16 + r + hv * 8)) * HID_ + h * HD_;
#pragma unroll
      for (int d = 0; d < 4; ++d)
        Ctx[row + d * 16 + ln] = f32_to_bf16(acc[qt][d][r] * inv);
    }
  }
}

// ---- kernel 5: output projection GEMM (16x64 tile per wave, +bias) ---------

__global__ __launch_bounds__(256) void k_gemm_out(const unsigned short* __restrict__ Ctx,
                                                  const unsigned short* __restrict__ Wp,
                                                  const float* __restrict__ bias,
                                                  float* __restrict__ out) {
  const int lane = threadIdx.x & 31;
  const int wave = threadIdx.x >> 5;
  const int tid  = blockIdx.x * 8 + wave;
  const int mt   = tid / (HID_ / 64);          // 0..511
  const int ntg  = tid % (HID_ / 64);          // 0..15
  const int hv   = lane >> 4;
  const int ln   = lane & 15;

  const unsigned short* arow = Ctx + (size_t)(mt * 16 + ln) * HID_;

  v8f acc0 = {}, acc1 = {}, acc2 = {}, acc3 = {};
#pragma unroll 2
  for (int kb = 0; kb < HID_ / 32; ++kb) {
    const int k0 = kb * 32 + hv * 8;
    const unsigned short* bb =
        Wp + (((size_t)kb * HID_ + ntg * 64 + ln) << 5) + hv * 8;
    v16bf a  = load_frag(arow + k0,      arow + k0 + 16);
    v16bf b0 = load_frag(bb,             bb + 16);
    v16bf b1 = load_frag(bb + 16 * 32,   bb + 16 * 32 + 16);
    v16bf b2 = load_frag(bb + 32 * 32,   bb + 32 * 32 + 16);
    v16bf b3 = load_frag(bb + 48 * 32,   bb + 48 * 32 + 16);
    if (kb + 1 < HID_ / 32) {
      __builtin_prefetch(arow + k0 + 32, 0, 0);
      __builtin_prefetch(bb + (HID_ << 5), 0, 0);
    }
    acc0 = wmma_bf16(a, b0, acc0);
    acc1 = wmma_bf16(a, b1, acc1);
    acc2 = wmma_bf16(a, b2, acc2);
    acc3 = wmma_bf16(a, b3, acc3);
  }

  v8f* accs[4] = {&acc0, &acc1, &acc2, &acc3};
#pragma unroll
  for (int t = 0; t < 4; ++t) {
    const int n  = ntg * 64 + t * 16 + ln;
    const float bv = bias[n];
#pragma unroll
    for (int r = 0; r < 8; ++r) {
      const int m = r + hv * 8;
      out[(size_t)(mt * 16 + m) * HID_ + n] = (*accs[t])[r] + bv;
    }
  }
}

// ---- host launcher ---------------------------------------------------------

extern "C" void kernel_launch(void* const* d_in, const int* in_sizes, int n_in,
                              void* d_out, int out_size, void* d_ws, size_t ws_size,
                              hipStream_t stream) {
  const float* x     = (const float*)d_in[0];
  // d_in[1] = causal mask (bool) — causality is applied analytically.
  const float* w_qkv = (const float*)d_in[2];
  const float* b_qkv = (const float*)d_in[3];
  const float* w_out = (const float*)d_in[4];
  const float* b_out = (const float*)d_in[5];
  float* out = (float*)d_out;

  // workspace carve-up (bf16 stored as ushort), 256B-aligned sections
  char* ws = (char*)d_ws;
  size_t off = 0;
  auto carve = [&](size_t bytes) {
    void* p = ws + off;
    off += (bytes + 255) & ~(size_t)255;
    return p;
  };
  unsigned short* xb    = (unsigned short*)carve((size_t)TOK_ * HID_ * 2);   // 16 MB
  unsigned short* wqkvp = (unsigned short*)carve((size_t)HID_ * NQKV_ * 2);  //  6 MB
  unsigned short* woutp = (unsigned short*)carve((size_t)HID_ * HID_ * 2);   //  2 MB
  unsigned short* Q     = (unsigned short*)carve((size_t)TOK_ * HID_ * 2);   // 16 MB [BH,S,64]
  unsigned short* Km    = (unsigned short*)carve((size_t)TOK_ * HID_ * 2);   // 16 MB [BH,S,64]
  unsigned short* Vt    = (unsigned short*)carve((size_t)TOK_ * HID_ * 2);   // 16 MB [BH,64,S]
  unsigned short* Ctx   = (unsigned short*)carve((size_t)TOK_ * HID_ * 2);   // 16 MB [B,S,H*D]

  // 1) convert x to bf16
  {
    const int n = TOK_ * HID_;
    k_convert_bf16<<<(n + 255) / 256, 256, 0, stream>>>(x, xb, n);
  }
  // 2) pack weights into K-blocked bf16 layout
  {
    const int n = HID_ * NQKV_;
    k_pack_weight<<<(n + 255) / 256, 256, 0, stream>>>(w_qkv, wqkvp, HID_, NQKV_);
  }
  {
    const int n = HID_ * HID_;
    k_pack_weight<<<(n + 255) / 256, 256, 0, stream>>>(w_out, woutp, HID_, HID_);
  }
  // 3) QKV projection GEMM: 512 x 48 (16x64) tiles, 8 waves/block
  k_gemm_qkv<<<(TOK_ / 16) * (NQKV_ / 64) / 8, 256, 0, stream>>>(
      xb, wqkvp, b_qkv, Q, Km, Vt);
  // 4) flash attention: 4*16*64 = 4096 waves, 8 waves/block
  k_attn<<<(B_ * NH_ * (S_ / 32)) / 8, 256, 0, stream>>>(Q, Km, Vt, Ctx);
  // 5) output projection GEMM: 512 x 16 (16x64) tiles
  k_gemm_out<<<(TOK_ / 16) * (HID_ / 64) / 8, 256, 0, stream>>>(
      Ctx, woutp, b_out, out);
  (void)in_sizes; (void)n_in; (void)out_size; (void)ws_size;
}
